// PointNetEncoder_86981677678874
// MI455X (gfx1250) — matrix operand order repaired and checked by reading, not compiled, not run on hardware
//
#include <hip/hip_runtime.h>

typedef __attribute__((ext_vector_type(16))) _Float16 v16h;
typedef __attribute__((ext_vector_type(8)))  _Float16 v8h;
typedef __attribute__((ext_vector_type(8)))  float    v8f;
typedef __attribute__((ext_vector_type(4)))  float    v4f;

static constexpr int kB   = 8;
static constexpr int kN   = 8192;
static constexpr int kNS  = 32;
static constexpr int kBEV = 200;

// ---------------- utility kernels ----------------
__global__ void pn_fill_f32(float* __restrict__ p, float v, int n) {
  int i = blockIdx.x * blockDim.x + threadIdx.x;
  if (i < n) p[i] = v;
}
__global__ void pn_fill_i32(int* __restrict__ p, int v, int n) {
  int i = blockIdx.x * blockDim.x + threadIdx.x;
  if (i < n) p[i] = v;
}

// xyz (B,1,18,N) -> pts (B,N,18)
__global__ void pn_transpose(const float* __restrict__ x, float* __restrict__ pts, int n) {
  int e = blockIdx.x * blockDim.x + threadIdx.x;
  if (e >= n) return;
  int c  = e % 18;
  int bn = e / 18;
  int nn = bn % kN;
  int b  = bn / kN;
  pts[e] = x[((size_t)b * 18 + c) * kN + nn];
}

// weight prep: W (cout,cin) f32 -> Wh (cout,Kpad) f16, zero-padded in K
__global__ void pn_wprep(const float* __restrict__ W, int cin, int Kpad,
                         _Float16* __restrict__ Wh, int total) {
  int e = blockIdx.x * blockDim.x + threadIdx.x;
  if (e >= total) return;
  int k = e % Kpad;
  int o = e / Kpad;
  Wh[e] = (k < cin) ? (_Float16)W[(size_t)o * cin + k] : (_Float16)0.0f;
}

// ---------------- farthest point sampling ----------------
__global__ __launch_bounds__(256)
void pn_fps(const float* __restrict__ pts, int stride, int N, int npoint,
            int* __restrict__ outIdx) {
  __shared__ float s_dist[kN];
  __shared__ float s_rv[256];
  __shared__ int   s_ri[256];
  const int b = blockIdx.x;
  const int t = threadIdx.x;
  const float* base = pts + (size_t)b * N * stride;
  for (int i = t; i < N; i += blockDim.x) s_dist[i] = 1e10f;
  __syncthreads();
  int far = 0;
  for (int it = 0; it < npoint; ++it) {
    if (t == 0) outIdx[b * npoint + it] = far;
    float cx = base[(size_t)far * stride + 0];
    float cy = base[(size_t)far * stride + 1];
    float cz = base[(size_t)far * stride + 2];
    float bv = -1.0f; int bi = 0;
    for (int i = t; i < N; i += blockDim.x) {
      float dx = base[(size_t)i * stride + 0] - cx;
      float dy = base[(size_t)i * stride + 1] - cy;
      float dz = base[(size_t)i * stride + 2] - cz;
      float d  = dx * dx + dy * dy + dz * dz;
      float o  = s_dist[i];
      d = d < o ? d : o;
      s_dist[i] = d;
      if (d > bv) { bv = d; bi = i; }
    }
    s_rv[t] = bv; s_ri[t] = bi;
    __syncthreads();
    for (int s = 128; s > 0; s >>= 1) {
      if (t < s) {
        if (s_rv[t + s] > s_rv[t] ||
            (s_rv[t + s] == s_rv[t] && s_ri[t + s] < s_ri[t])) {
          s_rv[t] = s_rv[t + s]; s_ri[t] = s_ri[t + s];
        }
      }
      __syncthreads();
    }
    far = s_ri[0];
    __syncthreads();
  }
}

// gather sampled coords: dst (B,P,3)
__global__ void pn_gather_xyz(const float* __restrict__ src, int stride, int N,
                              const int* __restrict__ fidx, int P,
                              float* __restrict__ dst, int total) {
  int e = blockIdx.x * blockDim.x + threadIdx.x;
  if (e >= total) return;
  int k  = e % 3;
  int bp = e / 3;
  int b  = bp / P;
  int id = fidx[bp];
  dst[e] = src[((size_t)b * N + id) * stride + k];
}

// ---------------- ball query ----------------
__global__ void pn_ball_query(const float* __restrict__ xyz, int stride, int N,
                              const float* __restrict__ nxyz, int P,
                              float r2, int ns, int* __restrict__ gidx, int total) {
  int e = blockIdx.x * blockDim.x + threadIdx.x;
  if (e >= total) return;
  int b = e / P;
  const float* q    = nxyz + (size_t)e * 3;
  const float* base = xyz + (size_t)b * N * stride;
  int* out = gidx + (size_t)e * ns;
  int cnt = 0, first = 0;
  for (int j = 0; j < N && cnt < ns; ++j) {
    float dx = base[(size_t)j * stride + 0] - q[0];
    float dy = base[(size_t)j * stride + 1] - q[1];
    float dz = base[(size_t)j * stride + 2] - q[2];
    float d2 = dx * dx + dy * dy + dz * dz;
    if (d2 <= r2) { if (cnt == 0) first = j; out[cnt++] = j; }
  }
  for (; cnt < ns; ++cnt) out[cnt] = first;
}

// grouped feature rows: X[(b,p,s)][0:3]=rel coords, [3:3+C]=feats, rest zero-pad
__global__ void pn_group(const float* __restrict__ xyz, int xs, int N,
                         const float* __restrict__ nxyz,
                         const float* __restrict__ feat, int fs, int C,
                         const int* __restrict__ gidx, int P, int ns,
                         float* __restrict__ Xb, int Kpad, int total) {
  int e = blockIdx.x * blockDim.x + threadIdx.x;
  if (e >= total) return;
  int k   = e % Kpad;
  int row = e / Kpad;
  int s   = row % ns;
  int bp  = row / ns;
  int b   = bp / P;
  int g   = gidx[(size_t)bp * ns + s];
  float v = 0.0f;
  if (k < 3)          v = xyz[((size_t)b * N + g) * xs + k] - nxyz[(size_t)bp * 3 + k];
  else if (k < 3 + C) v = feat[((size_t)b * N + g) * fs + (k - 3)];
  Xb[e] = v;
}

// ---------------- fused WMMA linear: Y = relu(g*(X·Wt + b) + be) ----------------
// X: (M,Kpad) f32 row-major, zero-padded. Wh: (cout,Kpad) f16, zero-padded.
// One wave per 16x(16*TN) output tile; A loaded once per K-step (b128 loads,
// packed f32->f16 convert), reused for TN V_WMMA_F32_16X16X32_F16 ops.
template <int TN>
__global__ __launch_bounds__(32)
void pn_wmma_linear(const float* __restrict__ X, int Kpad,
                    const _Float16* __restrict__ Wh,
                    const float* __restrict__ bias,
                    const float* __restrict__ gamma,
                    const float* __restrict__ beta,
                    float* __restrict__ Y, int ldy) {
  const int lane = threadIdx.x & 31;
  const int half = lane >> 4;   // lane half selects K sub-range per ISA layout
  const int l15  = lane & 15;
  const int tm   = blockIdx.x * 16;
  const int tn   = blockIdx.y * (16 * TN);

  const float* xrow = X + (size_t)(tm + l15) * Kpad + (half << 3);
  const _Float16* wrow[TN];
#pragma unroll
  for (int t = 0; t < TN; ++t)
    wrow[t] = Wh + (size_t)(tn + t * 16 + l15) * Kpad + (half << 3);

  v8f acc[TN];
#pragma unroll
  for (int t = 0; t < TN; ++t) acc[t] = (v8f){};

  for (int k0 = 0; k0 < Kpad; k0 += 32) {
    // A fragment: K = k0 + half*8 + {0..7} and k0 + 16 + half*8 + {0..7}
    v4f x0 = *(const v4f*)(xrow + k0);
    v4f x1 = *(const v4f*)(xrow + k0 + 4);
    v4f x2 = *(const v4f*)(xrow + k0 + 16);
    v4f x3 = *(const v4f*)(xrow + k0 + 20);
    v16h a;
#pragma unroll
    for (int e = 0; e < 4; ++e) {
      a[e]      = (_Float16)x0[e];
      a[e + 4]  = (_Float16)x1[e];
      a[e + 8]  = (_Float16)x2[e];
      a[e + 12] = (_Float16)x3[e];
    }
#pragma unroll
    for (int t = 0; t < TN; ++t) {
      // B fragment: K = k0 + half*8 + {0..7} and k0 + 16 + half*8 + {0..7}, column = tn + t*16 + (lane&15)
      v8h b0 = *(const v8h*)(wrow[t] + k0);
      v8h b1 = *(const v8h*)(wrow[t] + k0 + 16);
      v16h b = __builtin_shufflevector(b0, b1, 0, 1, 2, 3, 4, 5, 6, 7,
                                       8, 9, 10, 11, 12, 13, 14, 15);
      acc[t] = __builtin_amdgcn_wmma_f32_16x16x32_f16(false, a, false, b,
                                                      (short)0, acc[t], false, false);
    }
  }

#pragma unroll
  for (int t = 0; t < TN; ++t) {
    const int col = tn + t * 16 + l15;
    const float g  = gamma[col];
    const float bb = bias[col];
    const float be = beta[col];
#pragma unroll
    for (int r = 0; r < 8; ++r) {
      int row = tm + half * 8 + r;   // C/D layout: vgpr r -> M = half*8 + r
      float v = g * (acc[t][r] + bb) + be;
      Y[(size_t)row * ldy + col] = v > 0.0f ? v : 0.0f;
    }
  }
}

// max over the nsample axis: rows of Y are (bp*ns + s)
__global__ void pn_maxpool(const float* __restrict__ Y, int ld, int ns, int C,
                           float* __restrict__ out, int total) {
  int e = blockIdx.x * blockDim.x + threadIdx.x;
  if (e >= total) return;
  int c  = e % C;
  int bp = e / C;
  const float* p = Y + (size_t)bp * ns * ld + c;
  float m = p[0];
  for (int s = 1; s < ns; ++s) { float v = p[(size_t)s * ld]; m = v > m ? v : m; }
  out[e] = m;
}

// ---------------- 3-NN inverse-distance interpolation ----------------
__global__ void pn_nn3(const float* __restrict__ xyz1, int s1, int M1,
                       const float* __restrict__ xyz2, int s2, int M2,
                       int* __restrict__ nidx, float* __restrict__ nw, int total) {
  int e = blockIdx.x * blockDim.x + threadIdx.x;
  if (e >= total) return;
  int b = e / M1;
  const float* q    = xyz1 + (size_t)e * s1;
  const float* base = xyz2 + (size_t)b * M2 * s2;
  float d0 = 3e38f, d1 = 3e38f, d2 = 3e38f;
  int   i0 = 0, i1 = 0, i2 = 0;
  for (int j = 0; j < M2; ++j) {
    float dx = base[(size_t)j * s2 + 0] - q[0];
    float dy = base[(size_t)j * s2 + 1] - q[1];
    float dz = base[(size_t)j * s2 + 2] - q[2];
    float d  = dx * dx + dy * dy + dz * dz;
    if (d < d0)      { d2 = d1; i2 = i1; d1 = d0; i1 = i0; d0 = d; i0 = j; }
    else if (d < d1) { d2 = d1; i2 = i1; d1 = d;  i1 = j; }
    else if (d < d2) { d2 = d;  i2 = j; }
  }
  float w0 = 1.0f / (d0 + 1e-8f);
  float w1 = 1.0f / (d1 + 1e-8f);
  float w2 = 1.0f / (d2 + 1e-8f);
  float s  = w0 + w1 + w2;
  nidx[(size_t)e * 3 + 0] = i0; nidx[(size_t)e * 3 + 1] = i1; nidx[(size_t)e * 3 + 2] = i2;
  nw[(size_t)e * 3 + 0] = w0 / s; nw[(size_t)e * 3 + 1] = w1 / s; nw[(size_t)e * 3 + 2] = w2 / s;
}

// X rows: [0:C1]=points1, [C1:C1+C2]=interp(points2), rest zero-pad
__global__ void pn_fp_buildx(const float* __restrict__ p1, int ld1, int C1,
                             const float* __restrict__ p2, int ld2, int C2,
                             const int* __restrict__ nidx, const float* __restrict__ nw,
                             int M1, int M2, int Kpad,
                             float* __restrict__ Xb, int total) {
  int e = blockIdx.x * blockDim.x + threadIdx.x;
  if (e >= total) return;
  int k = e % Kpad;
  int r = e / Kpad;           // r = b*M1 + i
  int b = r / M1;
  float v = 0.0f;
  if (k < C1) {
    v = p1[(size_t)r * ld1 + k];
  } else if (k < C1 + C2) {
    int c = k - C1;
    const int*   ni = nidx + (size_t)r * 3;
    const float* w  = nw + (size_t)r * 3;
    const float* pb = p2 + (size_t)b * M2 * ld2;
    v = w[0] * pb[(size_t)ni[0] * ld2 + c] +
        w[1] * pb[(size_t)ni[1] * ld2 + c] +
        w[2] * pb[(size_t)ni[2] * ld2 + c];
  }
  Xb[e] = v;
}

// ---------------- deterministic BEV scatter (max-n wins) ----------------
__global__ void pn_scatter_winner(const float* __restrict__ pts,
                                  int* __restrict__ winner, int total) {
  int e = blockIdx.x * blockDim.x + threadIdx.x;
  if (e >= total) return;
  int b = e / kN, n = e % kN;
  int ix = (int)floorf(pts[(size_t)e * 18 + 0]);
  int iy = (int)floorf(pts[(size_t)e * 18 + 1]);
  if (ix < 0 || ix >= kBEV || iy < 0 || iy >= kBEV) return;
  atomicMax(&winner[((size_t)b * kBEV + iy) * kBEV + ix], n);
}

__global__ void pn_scatter_write(const float* __restrict__ pts,
                                 const float* __restrict__ l0p,
                                 const int* __restrict__ winner,
                                 float* __restrict__ out, int total) {
  int e = blockIdx.x * blockDim.x + threadIdx.x;
  if (e >= total) return;
  int c  = e & 127;
  int bn = e >> 7;
  int b  = bn / kN, n = bn % kN;
  int ix = (int)floorf(pts[(size_t)bn * 18 + 0]);
  int iy = (int)floorf(pts[(size_t)bn * 18 + 1]);
  if (ix < 0 || ix >= kBEV || iy < 0 || iy >= kBEV) return;
  if (winner[((size_t)b * kBEV + iy) * kBEV + ix] != n) return;
  out[(((size_t)b * 128 + c) * kBEV + iy) * kBEV + ix] = l0p[(size_t)bn * 128 + c];
}

// ---------------- host orchestration ----------------
extern "C" void kernel_launch(void* const* d_in, const int* in_sizes, int n_in,
                              void* d_out, int out_size, void* d_ws, size_t ws_size,
                              hipStream_t stream) {
  struct Layer { const float *W, *b, *g, *be; int cout, cin; };
  // logical order: sa1(0-2) sa2(3-5) sa3(6-8) fp3(9-10) fp2(11-12) fp1(13-15)
  static const int dims[16][2] = {
    {32, 21}, {32, 32}, {64, 32},
    {64, 67}, {64, 64}, {128, 64},
    {128, 131}, {128, 128}, {256, 128},
    {256, 384}, {256, 256},
    {128, 320}, {128, 128},
    {128, 128}, {128, 128}, {128, 128}};
  Layer L[16];
  const float* xin = nullptr;
  bool xyz_first = (n_in > 0 && in_sizes[0] == kB * 18 * kN);
  if (xyz_first) {
    // insertion order: xyz, then sa1..fp1, each layer W,b,g,be
    xin = (const float*)d_in[0];
    int p = 1;
    for (int i = 0; i < 16; ++i) {
      L[i].W  = (const float*)d_in[p + 0];
      L[i].b  = (const float*)d_in[p + 1];
      L[i].g  = (const float*)d_in[p + 2];
      L[i].be = (const float*)d_in[p + 3];
      L[i].cout = dims[i][0]; L[i].cin = dims[i][1];
      p += 4;
    }
  } else {
    // pytree-sorted order: fp1,fp2,fp3,sa1,sa2,sa3; each layer W,b,be,g; xyz last
    static const int order[16] = {13, 14, 15, 11, 12, 9, 10, 0, 1, 2, 3, 4, 5, 6, 7, 8};
    int p = 0;
    for (int oi = 0; oi < 16; ++oi) {
      int i = order[oi];
      L[i].W  = (const float*)d_in[p + 0];
      L[i].b  = (const float*)d_in[p + 1];
      L[i].be = (const float*)d_in[p + 2];
      L[i].g  = (const float*)d_in[p + 3];
      L[i].cout = dims[i][0]; L[i].cin = dims[i][1];
      p += 4;
    }
    xin = (const float*)d_in[64];
  }

  // workspace bump allocator
  char* wbase = (char*)d_ws;
  size_t off = 0;
  auto alloc = [&](size_t bytes) -> void* {
    off = (off + 255) & ~(size_t)255;
    void* p = wbase + off;
    off += bytes;
    return p;
  };
  float* pts    = (float*)alloc((size_t)kB * kN * 18 * 4);
  float* l1x    = (float*)alloc((size_t)kB * 256 * 3 * 4);
  float* l2x    = (float*)alloc((size_t)kB * 64 * 3 * 4);
  float* l3x    = (float*)alloc((size_t)kB * 16 * 3 * 4);
  float* l1p    = (float*)alloc((size_t)kB * 256 * 64 * 4);
  float* l2p    = (float*)alloc((size_t)kB * 64 * 128 * 4);
  float* l3p    = (float*)alloc((size_t)kB * 16 * 256 * 4);
  int*   fidx   = (int*)alloc((size_t)kB * 256 * 4);
  int*   gidx   = (int*)alloc((size_t)kB * 256 * kNS * 4);
  int*   nnidx  = (int*)alloc((size_t)kB * kN * 3 * 4);
  float* nnw    = (float*)alloc((size_t)kB * kN * 3 * 4);
  int*   winner = (int*)alloc((size_t)kB * kBEV * kBEV * 4);
  float* bigA   = (float*)alloc((size_t)kB * kN * 128 * 4);
  float* bigB   = (float*)alloc((size_t)kB * kN * 128 * 4);
  (void)ws_size;

  auto g1 = [](int n) { return dim3((unsigned)((n + 255) / 256)); };

  // per-layer padded f16 weights
  _Float16* Wh[16];
  int kpads[16];
  for (int i = 0; i < 16; ++i) {
    int kp = (L[i].cin + 31) & ~31;
    kpads[i] = kp;
    Wh[i] = (_Float16*)alloc((size_t)L[i].cout * kp * 2);
    int t = L[i].cout * kp;
    pn_wprep<<<g1(t), 256, 0, stream>>>(L[i].W, L[i].cin, kp, Wh[i], t);
  }

  auto linear = [&](const float* X, int li, float* Y, int M) {
    int cout = L[li].cout;
    int kp   = kpads[li];
    if ((cout & 63) == 0) {
      dim3 grid((unsigned)(M / 16), (unsigned)(cout / 64));
      pn_wmma_linear<4><<<grid, 32, 0, stream>>>(X, kp, Wh[li], L[li].b, L[li].g,
                                                 L[li].be, Y, cout);
    } else {
      dim3 grid((unsigned)(M / 16), (unsigned)(cout / 32));
      pn_wmma_linear<2><<<grid, 32, 0, stream>>>(X, kp, Wh[li], L[li].b, L[li].g,
                                                 L[li].be, Y, cout);
    }
  };

  // transpose (B,18,N) -> (B,N,18)
  { int t = kB * kN * 18; pn_transpose<<<g1(t), 256, 0, stream>>>(xin, pts, t); }

  // -------- SA1: 8192 -> 256, r=0.1, MLP 21->32->32->64 --------
  pn_fps<<<kB, 256, 0, stream>>>(pts, 18, kN, 256, fidx);
  { int t = kB * 256 * 3; pn_gather_xyz<<<g1(t), 256, 0, stream>>>(pts, 18, kN, fidx, 256, l1x, t); }
  { int t = kB * 256;     pn_ball_query<<<g1(t), 256, 0, stream>>>(pts, 18, kN, l1x, 256, 0.01f, kNS, gidx, t); }
  { int t = kB * 256 * kNS * 32;
    pn_group<<<g1(t), 256, 0, stream>>>(pts, 18, kN, l1x, pts, 18, 18, gidx, 256, kNS, bigA, 32, t); }
  const int M1 = kB * 256 * kNS;  // 65536
  linear(bigA, 0, bigB, M1);
  linear(bigB, 1, bigA, M1);
  linear(bigA, 2, bigB, M1);
  { int t = kB * 256 * 64; pn_maxpool<<<g1(t), 256, 0, stream>>>(bigB, 64, kNS, 64, l1p, t); }

  // -------- SA2: 256 -> 64, r=0.2, MLP 67->64->64->128 --------
  pn_fps<<<kB, 256, 0, stream>>>(l1x, 3, 256, 64, fidx);
  { int t = kB * 64 * 3; pn_gather_xyz<<<g1(t), 256, 0, stream>>>(l1x, 3, 256, fidx, 64, l2x, t); }
  { int t = kB * 64;     pn_ball_query<<<g1(t), 256, 0, stream>>>(l1x, 3, 256, l2x, 64, 0.04f, kNS, gidx, t); }
  { int t = kB * 64 * kNS * 96;
    pn_group<<<g1(t), 256, 0, stream>>>(l1x, 3, 256, l2x, l1p, 64, 64, gidx, 64, kNS, bigA, 96, t); }
  const int M2 = kB * 64 * kNS;  // 16384
  linear(bigA, 3, bigB, M2);
  linear(bigB, 4, bigA, M2);
  linear(bigA, 5, bigB, M2);
  { int t = kB * 64 * 128; pn_maxpool<<<g1(t), 256, 0, stream>>>(bigB, 128, kNS, 128, l2p, t); }

  // -------- SA3: 64 -> 16, r=0.4, MLP 131->128->128->256 --------
  pn_fps<<<kB, 256, 0, stream>>>(l2x, 3, 64, 16, fidx);
  { int t = kB * 16 * 3; pn_gather_xyz<<<g1(t), 256, 0, stream>>>(l2x, 3, 64, fidx, 16, l3x, t); }
  { int t = kB * 16;     pn_ball_query<<<g1(t), 256, 0, stream>>>(l2x, 3, 64, l3x, 16, 0.16f, kNS, gidx, t); }
  { int t = kB * 16 * kNS * 160;
    pn_group<<<g1(t), 256, 0, stream>>>(l2x, 3, 64, l3x, l2p, 128, 128, gidx, 16, kNS, bigA, 160, t); }
  const int M3 = kB * 16 * kNS;  // 4096
  linear(bigA, 6, bigB, M3);
  linear(bigB, 7, bigA, M3);
  linear(bigA, 8, bigB, M3);
  { int t = kB * 16 * 256; pn_maxpool<<<g1(t), 256, 0, stream>>>(bigB, 256, kNS, 256, l3p, t); }

  // -------- FP3: interp l3p -> l2, concat l2p -> MLP 384->256->256 --------
  { int t = kB * 64; pn_nn3<<<g1(t), 256, 0, stream>>>(l2x, 3, 64, l3x, 3, 16, nnidx, nnw, t); }
  { int t = kB * 64 * 384;
    pn_fp_buildx<<<g1(t), 256, 0, stream>>>(l2p, 128, 128, l3p, 256, 256, nnidx, nnw, 64, 16, 384, bigA, t); }
  const int Mf3 = kB * 64;  // 512
  linear(bigA, 9,  bigB, Mf3);
  linear(bigB, 10, bigA, Mf3);   // new l2p (512 x 256) lives in bigA

  // -------- FP2: interp -> l1, concat l1p -> MLP 320->128->128 --------
  { int t = kB * 256; pn_nn3<<<g1(t), 256, 0, stream>>>(l1x, 3, 256, l2x, 3, 64, nnidx, nnw, t); }
  { int t = kB * 256 * 320;
    pn_fp_buildx<<<g1(t), 256, 0, stream>>>(l1p, 64, 64, bigA, 256, 256, nnidx, nnw, 256, 64, 320, bigB, t); }
  const int Mf2 = kB * 256;  // 2048
  linear(bigB, 11, bigA, Mf2);
  linear(bigA, 12, bigB, Mf2);   // new l1p (2048 x 128) lives in bigB

  // -------- FP1: interp -> 8192 pts -> MLP 128->128->128->128 --------
  { int t = kB * kN; pn_nn3<<<g1(t), 256, 0, stream>>>(pts, 18, kN, l1x, 3, 256, nnidx, nnw, t); }
  { int t = kB * kN * 128;
    pn_fp_buildx<<<g1(t), 256, 0, stream>>>(nullptr, 0, 0, bigB, 128, 128, nnidx, nnw, kN, 256, 128, bigA, t); }
  const int Mf1 = kB * kN;  // 65536
  linear(bigA, 13, bigB, Mf1);
  linear(bigB, 14, bigA, Mf1);
  linear(bigA, 15, bigB, Mf1);   // l0p (65536 x 128) lives in bigB

  // -------- deterministic BEV scatter --------
  { int t = kB * kBEV * kBEV; pn_fill_i32<<<g1(t), 256, 0, stream>>>(winner, -1, t); }
  { int t = kB * kN;          pn_scatter_winner<<<g1(t), 256, 0, stream>>>(pts, winner, t); }
  { int t = out_size;         pn_fill_f32<<<g1(t), 256, 0, stream>>>((float*)d_out, 0.0f, t); }
  { int t = kB * kN * 128;    pn_scatter_write<<<g1(t), 256, 0, stream>>>(pts, bigB, winner, (float*)d_out, t); }
}
